// LinearAttention_32066225832173
// MI455X (gfx1250) — matrix operand order repaired and checked by reading, not compiled, not run on hardware
//
#include <hip/hip_runtime.h>
#include <hip/hip_bf16.h>
#include <stdint.h>

#define B_ 4
#define N_ 4096
#define D_ 1024
#define H_ 16
#define HD_ 64

typedef __bf16 bf16;
typedef __attribute__((ext_vector_type(16))) __bf16 v16bf;
typedef __attribute__((ext_vector_type(8)))  __bf16 bf16x8;
typedef __attribute__((ext_vector_type(4)))  __bf16 bf16x4;
typedef __attribute__((ext_vector_type(8)))  float  v8f;
typedef __attribute__((ext_vector_type(4)))  float  f32x4;
typedef __attribute__((ext_vector_type(4)))  unsigned int u32x4;
typedef __attribute__((ext_vector_type(8)))  int    i32x8;
typedef __attribute__((ext_vector_type(4)))  int    i32x4;

#if __has_builtin(__builtin_amdgcn_tensor_load_to_lds) && __has_builtin(__builtin_amdgcn_s_wait_tensorcnt)
#define USE_TDM 1
#else
#define USE_TDM 0
#endif

static __device__ __forceinline__ v8f v8f_zero() {
  v8f z;
#pragma unroll
  for (int i = 0; i < 8; ++i) z[i] = 0.f;
  return z;
}

// ---- WMMA fragment builders (wave32, v_wmma_f32_16x16x32_bf16) ------------
// A 16x32 (MxK), source row-major s[m*stride + k] -> two ds_load_b128.
static __device__ __forceinline__ v16bf frag_a_rm(const bf16* s, int stride, int lane) {
  const int row = lane & 15;
  const int kq  = (lane >> 4) << 3;
  const bf16* p = s + row * stride + kq;
  bf16x8 lo = *(const bf16x8*)(p);
  bf16x8 hi = *(const bf16x8*)(p + 16);
  return __builtin_shufflevector(lo, hi, 0, 1, 2, 3, 4, 5, 6, 7,
                                 8, 9, 10, 11, 12, 13, 14, 15);
}

// B 32x16 (KxN), source n-major s[n*stride + k] -> two ds_load_b128.
static __device__ __forceinline__ v16bf frag_b_nk(const bf16* s, int stride, int lane) {
  const int col = lane & 15;
  const int kq  = (lane >> 4) << 3;
  const bf16* p = s + col * stride + kq;
  bf16x8 lo = *(const bf16x8*)(p);
  bf16x8 hi = *(const bf16x8*)(p + 16);
  return __builtin_shufflevector(lo, hi, 0, 1, 2, 3, 4, 5, 6, 7,
                                 8, 9, 10, 11, 12, 13, 14, 15);
}

static __device__ __forceinline__ v8f wmma_bf16(v16bf a, v16bf b, v8f c) {
  return __builtin_amdgcn_wmma_f32_16x16x32_bf16(false, a, false, b, (short)0, c, false, false);
}

static __device__ __forceinline__ float elu1(float x) {
  return x > 0.f ? x + 1.f : __expf(x);
}

#if USE_TDM
// ---- Tensor Data Mover: 2D bf16 tile -> LDS with row padding --------------
static __device__ __forceinline__ void tdm_load_tile(
    const void* gsrc, unsigned int lds_off,
    unsigned int tile_w, unsigned int tile_h,      // elements / rows
    unsigned int tens_w, unsigned int tens_h,      // elements
    unsigned int row_stride_elems,
    unsigned int pad_interval_code, unsigned int pad_amount_code) {
  const unsigned long long ga = (unsigned long long)(uintptr_t)gsrc;
  u32x4 g0;
  g0[0] = 1u;                                      // count=1, user descriptor
  g0[1] = lds_off;                                 // lds_addr
  g0[2] = (unsigned int)ga;                        // global_addr[31:0]
  g0[3] = (unsigned int)((ga >> 32) & 0x01FFFFFFu) | (2u << 30);  // addr[56:32], type=2
  i32x8 g1;
  g1[0] = (int)((1u << 16)                                        // data_size = 2B
                | (1u << 20)                                      // pad_enable
                | (pad_interval_code << 22) | (pad_amount_code << 25));
  g1[1] = (int)((tens_w & 0xFFFFu) << 16);                        // tensor_dim0[15:0]
  g1[2] = (int)((tens_w >> 16) | ((tens_h & 0xFFFFu) << 16));     // d0[31:16] | d1[15:0]
  g1[3] = (int)((tens_h >> 16) | (tile_w << 16));                 // d1[31:16] | tile_dim0
  g1[4] = (int)(tile_h & 0xFFFFu);                                // tile_dim1 (tile_dim2=0)
  g1[5] = (int)row_stride_elems;                                  // dim0_stride[31:0]
  g1[6] = 0;
  g1[7] = 0;
  const i32x4 z4 = {0, 0, 0, 0};
#if defined(__clang_major__) && (__clang_major__ >= 23)
  const i32x8 z8 = {0, 0, 0, 0, 0, 0, 0, 0};
  __builtin_amdgcn_tensor_load_to_lds(g0, g1, z4, z4, z8, 0);
#else
  __builtin_amdgcn_tensor_load_to_lds(g0, g1, z4, z4, 0);
#endif
}

static __device__ __forceinline__ unsigned int lds_off_of(const void* p) {
  return (unsigned int)(uintptr_t)p;   // low 32 bits of flat addr == LDS offset
}
#endif  // USE_TDM

// ---- fp32 -> bf16 convert (vectorized) ------------------------------------
__global__ void cvt_kernel(const float* __restrict__ in, bf16* __restrict__ out, size_t n) {
  size_t i = (size_t)blockIdx.x * blockDim.x + threadIdx.x;
  const size_t stride = (size_t)gridDim.x * blockDim.x;
  const size_t n4 = n >> 2;
  for (; i < n4; i += stride) {
    f32x4 v = ((const f32x4*)in)[i];
    bf16x4 o;
#pragma unroll
    for (int j = 0; j < 4; ++j) o[j] = (bf16)v[j];
    ((bf16x4*)out)[i] = o;
  }
}

// ---- big GEMM: C[M,Nout] = A[M,K] @ W[Nout,K]^T ---------------------------
// 256x128 block tile, BK=64, 8 waves (4x2), 64x64 per wave -> 32 WMMA/barrier.
// MODE 0: epilogue splits q/k/v, elu+1 on q,k, bf16 out.
// MODE 1: epilogue adds bias, fp32 out.
template <int MODE>
static __device__ __forceinline__ void gemm_body(
    const bf16* __restrict__ A, const bf16* __restrict__ W, int K, int Mtot, int Ntot,
    bf16* __restrict__ qo, bf16* __restrict__ ko, bf16* __restrict__ vo,
    const float* __restrict__ bias, float* __restrict__ fo) {
  const int tid = threadIdx.x, lane = tid & 31, w = tid >> 5;
  const int gm0 = blockIdx.x * 256, gn0 = blockIdx.y * 128;
  const int wm = w >> 1, wn = w & 1;
  constexpr int LDT = 72;                          // 64 data + 8 pad bf16 per row
  __shared__ alignas(128) bf16 As[2][256 * LDT];
  __shared__ alignas(128) bf16 Bs[2][128 * LDT];

  v8f acc[4][4];
#pragma unroll
  for (int i = 0; i < 4; ++i)
#pragma unroll
    for (int j = 0; j < 4; ++j) acc[i][j] = v8f_zero();

#if USE_TDM
  // pad: 64-elem (128B) rows -> interval code 4 (16x8B), amount code 3 (4 DWORDs)
  if (w == 0)
    tdm_load_tile(A + (size_t)gm0 * K, lds_off_of(&As[0][0]), 64, 256,
                  (unsigned)K, (unsigned)Mtot, (unsigned)K, 4, 3);
  if (w == 1)
    tdm_load_tile(W + (size_t)gn0 * K, lds_off_of(&Bs[0][0]), 64, 128,
                  (unsigned)K, (unsigned)Ntot, (unsigned)K, 4, 3);
  if (w < 2) __builtin_amdgcn_s_wait_tensorcnt(0);
  __syncthreads();

  for (int kb = 0, it = 0; kb < K; kb += 64, ++it) {
    const int cur = it & 1, nxt = cur ^ 1;
    const bool more = (kb + 64) < K;
    if (more) {
      if (w == 0)
        tdm_load_tile(A + (size_t)gm0 * K + kb + 64, lds_off_of(&As[nxt][0]), 64, 256,
                      (unsigned)K, (unsigned)Mtot, (unsigned)K, 4, 3);
      if (w == 1)
        tdm_load_tile(W + (size_t)gn0 * K + kb + 64, lds_off_of(&Bs[nxt][0]), 64, 128,
                      (unsigned)K, (unsigned)Ntot, (unsigned)K, 4, 3);
    }
#pragma unroll
    for (int ks = 0; ks < 64; ks += 32) {
      v16bf fa[4], fb[4];
#pragma unroll
      for (int mt = 0; mt < 4; ++mt)
        fa[mt] = frag_a_rm(&As[cur][(wm * 64 + mt * 16) * LDT + ks], LDT, lane);
#pragma unroll
      for (int nt = 0; nt < 4; ++nt)
        fb[nt] = frag_b_nk(&Bs[cur][(wn * 64 + nt * 16) * LDT + ks], LDT, lane);
#pragma unroll
      for (int mt = 0; mt < 4; ++mt)
#pragma unroll
        for (int nt = 0; nt < 4; ++nt)
          acc[mt][nt] = wmma_bf16(fa[mt], fb[nt], acc[mt][nt]);
    }
    if (more && w < 2) __builtin_amdgcn_s_wait_tensorcnt(0);
    __syncthreads();
  }
#else
  const int ar = tid >> 3, ac = (tid & 7) << 3;    // 32 rows x (8x8) cols per pass
  for (int kb = 0; kb < K; kb += 64) {
#pragma unroll
    for (int p = 0; p < 8; ++p) {
      const int r = ar + p * 32;
      *(bf16x8*)&As[0][r * LDT + ac] = *(const bf16x8*)&A[(size_t)(gm0 + r) * K + kb + ac];
    }
#pragma unroll
    for (int p = 0; p < 4; ++p) {
      const int r = ar + p * 32;
      *(bf16x8*)&Bs[0][r * LDT + ac] = *(const bf16x8*)&W[(size_t)(gn0 + r) * K + kb + ac];
    }
    __syncthreads();
#pragma unroll
    for (int ks = 0; ks < 64; ks += 32) {
      v16bf fa[4], fb[4];
#pragma unroll
      for (int mt = 0; mt < 4; ++mt)
        fa[mt] = frag_a_rm(&As[0][(wm * 64 + mt * 16) * LDT + ks], LDT, lane);
#pragma unroll
      for (int nt = 0; nt < 4; ++nt)
        fb[nt] = frag_b_nk(&Bs[0][(wn * 64 + nt * 16) * LDT + ks], LDT, lane);
#pragma unroll
      for (int mt = 0; mt < 4; ++mt)
#pragma unroll
        for (int nt = 0; nt < 4; ++nt)
          acc[mt][nt] = wmma_bf16(fa[mt], fb[nt], acc[mt][nt]);
    }
    __syncthreads();
  }
#endif

  const int lr = lane & 15, lhi = lane >> 4;
#pragma unroll
  for (int mt = 0; mt < 4; ++mt)
#pragma unroll
    for (int nt = 0; nt < 4; ++nt)
#pragma unroll
      for (int r = 0; r < 8; ++r) {
        const int gm = gm0 + wm * 64 + mt * 16 + r + (lhi << 3);
        const int gc = gn0 + wn * 64 + nt * 16 + lr;
        const float val = acc[mt][nt][r];
        if (MODE == 0) {
          const int seg = gc >> 10, cc = gc & 1023;
          const size_t o = (size_t)gm * D_ + cc;
          if (seg == 0)      qo[o] = (bf16)elu1(val);
          else if (seg == 1) ko[o] = (bf16)elu1(val);
          else               vo[o] = (bf16)val;
        } else {
          fo[(size_t)gm * D_ + gc] = val + bias[gc];
        }
      }
}

__global__ __launch_bounds__(256)
void gemm_qkv(const bf16* __restrict__ A, const bf16* __restrict__ W,
              bf16* __restrict__ qo, bf16* __restrict__ ko, bf16* __restrict__ vo) {
  gemm_body<0>(A, W, D_, B_ * N_, 3 * D_, qo, ko, vo, nullptr, nullptr);
}

__global__ __launch_bounds__(256)
void gemm_out(const bf16* __restrict__ A, const bf16* __restrict__ W,
              const float* __restrict__ bias, float* __restrict__ fo) {
  gemm_body<1>(A, W, D_, B_ * N_, D_, nullptr, nullptr, nullptr, bias, fo);
}

// ---- kv[b,h] = K^T V : transpose K,V chunks at staging, vector frag reads --
__global__ __launch_bounds__(256)
void kv_kernel(const bf16* __restrict__ kb, const bf16* __restrict__ vb,
               bf16* __restrict__ kvout) {
  const int tid = threadIdx.x, lane = tid & 31, w = tid >> 5;
  const int bh = blockIdx.x, b = bh >> 4, h = bh & 15;
  constexpr int LDK = 40;                     // 32 n + 8 pad
  __shared__ alignas(16) bf16 Kt[64 * LDK];   // [d][n]
  __shared__ alignas(16) bf16 Vt[64 * LDK];   // [m][n]
  v8f acc[2] = {v8f_zero(), v8f_zero()};

  const int lrow = tid >> 3, lcb = (tid & 7) << 3;
  const size_t base = ((size_t)b * N_) * D_ + h * HD_;
  for (int nb = 0; nb < N_; nb += 32) {
    bf16x8 kr = *(const bf16x8*)&kb[base + (size_t)(nb + lrow) * D_ + lcb];
    bf16x8 vr = *(const bf16x8*)&vb[base + (size_t)(nb + lrow) * D_ + lcb];
#pragma unroll
    for (int j = 0; j < 8; ++j) {
      Kt[(lcb + j) * LDK + lrow] = kr[j];
      Vt[(lcb + j) * LDK + lrow] = vr[j];
    }
    __syncthreads();
#pragma unroll
    for (int i = 0; i < 2; ++i) {
      const int t = w * 2 + i, tm = t >> 2, tn = t & 3;
      v16bf fa = frag_a_rm(&Kt[(tm * 16) * LDK], LDK, lane);   // A[d][n]
      v16bf fb = frag_b_nk(&Vt[(tn * 16) * LDK], LDK, lane);   // B[n][m]
      acc[i] = wmma_bf16(fa, fb, acc[i]);
    }
    __syncthreads();
  }

  const int lr = lane & 15, lhi = lane >> 4;
#pragma unroll
  for (int i = 0; i < 2; ++i) {
    const int t = w * 2 + i, tm = t >> 2, tn = t & 3;
#pragma unroll
    for (int r = 0; r < 8; ++r) {
      const int d = tm * 16 + r + (lhi << 3), m = tn * 16 + lr;
      kvout[((size_t)bh * HD_ + d) * HD_ + m] = (bf16)acc[i][r];
    }
  }
}

// ---- ksum[b,h,d] = sum_n k ------------------------------------------------
__global__ __launch_bounds__(64)
void ksum_kernel(const bf16* __restrict__ kb, float* __restrict__ ksum) {
  const int bh = blockIdx.x, b = bh >> 4, h = bh & 15, d = threadIdx.x;
  const bf16* p = kb + ((size_t)b * N_) * D_ + h * HD_ + d;
  float s = 0.f;
  for (int n = 0; n < N_; ++n) s += (float)p[(size_t)n * D_];
  ksum[bh * HD_ + d] = s;
}

// ---- z[b,n,h] = 1/(q . ksum + eps) ----------------------------------------
__global__ __launch_bounds__(256)
void z_kernel(const bf16* __restrict__ qb, const float* __restrict__ ksum,
              float* __restrict__ z) {
  const int idx = blockIdx.x * 256 + threadIdx.x;   // (b*N+n)*H + h
  const int bn = idx >> 4, h = idx & 15;
  const bf16* p = qb + (size_t)bn * D_ + h * HD_;
  const float* s = ksum + ((bn / N_) * H_ + h) * HD_;
  float acc = 0.f;
#pragma unroll
  for (int d = 0; d < HD_; ++d) acc += (float)p[d] * s[d];
  z[idx] = 1.f / (acc + 1e-6f);
}

// ---- attn = (q @ kv) * z ---------------------------------------------------
__global__ __launch_bounds__(256)
void attn_kernel(const bf16* __restrict__ qb, const bf16* __restrict__ kv,
                 const float* __restrict__ z, bf16* __restrict__ attn) {
  const int tid = threadIdx.x, lane = tid & 31, w = tid >> 5;
  const int bh = blockIdx.y, b = bh >> 4, h = bh & 15;
  const int n0 = blockIdx.x * 128;
  constexpr int LDV = 72;
  __shared__ alignas(16) bf16 KVt[64 * LDV];   // [m][d]
  {
    const int r = tid >> 2, cb = (tid & 3) << 4;
    const bf16* src = kv + (size_t)bh * HD_ * HD_;
    bf16x8 a0 = *(const bf16x8*)&src[r * HD_ + cb];
    bf16x8 a1 = *(const bf16x8*)&src[r * HD_ + cb + 8];
#pragma unroll
    for (int j = 0; j < 8; ++j) {
      KVt[(cb + j) * LDV + r]     = a0[j];
      KVt[(cb + 8 + j) * LDV + r] = a1[j];
    }
  }
  __syncthreads();

  v8f acc[4];
#pragma unroll
  for (int i = 0; i < 4; ++i) acc[i] = v8f_zero();

  const size_t qrow0 = ((size_t)b * N_ + n0 + w * 16) * D_ + h * HD_;
#pragma unroll
  for (int ks = 0; ks < HD_; ks += 32) {
    v16bf fa = frag_a_rm(&qb[qrow0 + ks], D_, lane);
#pragma unroll
    for (int nt = 0; nt < 4; ++nt) {
      v16bf fb = frag_b_nk(&KVt[(nt * 16) * LDV + ks], LDV, lane);
      acc[nt] = wmma_bf16(fa, fb, acc[nt]);
    }
  }

  const int lr = lane & 15, lhi = lane >> 4;
#pragma unroll
  for (int nt = 0; nt < 4; ++nt)
#pragma unroll
    for (int r = 0; r < 8; ++r) {
      const int n = n0 + w * 16 + r + (lhi << 3);
      const int m = nt * 16 + lr;
      const float zv = z[((size_t)b * N_ + n) * H_ + h];
      attn[((size_t)b * N_ + n) * D_ + h * HD_ + m] = (bf16)(acc[nt][r] * zv);
    }
}

extern "C" void kernel_launch(void* const* d_in, const int* in_sizes, int n_in,
                              void* d_out, int out_size, void* d_ws, size_t ws_size,
                              hipStream_t stream) {
  const float* x     = (const float*)d_in[0];
  const float* w_qkv = (const float*)d_in[1];
  const float* w_out = (const float*)d_in[2];
  const float* b_out = (const float*)d_in[3];
  float* out = (float*)d_out;

  char* p = (char*)d_ws;
  const size_t SZ = (size_t)B_ * N_ * D_;
  bf16*  qb   = (bf16*)p;  p += SZ * 2;
  bf16*  kbuf = (bf16*)p;  p += SZ * 2;
  bf16*  vbuf = (bf16*)p;  p += SZ * 2;
  bf16*  attn = (bf16*)p;  p += SZ * 2;
  bf16*  xb   = (bf16*)p;  p += SZ * 2;
  bf16*  wqb  = (bf16*)p;  p += (size_t)3 * D_ * D_ * 2;
  bf16*  wob  = (bf16*)p;  p += (size_t)D_ * D_ * 2;
  bf16*  kvb  = (bf16*)p;  p += (size_t)B_ * H_ * HD_ * HD_ * 2;
  float* ksum = (float*)p; p += (size_t)B_ * H_ * HD_ * 4;
  float* zbuf = (float*)p; p += (size_t)B_ * N_ * H_ * 4;

  cvt_kernel<<<2048, 256, 0, stream>>>(x, xb, SZ);
  cvt_kernel<<<1024, 256, 0, stream>>>(w_qkv, wqb, (size_t)3 * D_ * D_);
  cvt_kernel<<<512,  256, 0, stream>>>(w_out, wob, (size_t)D_ * D_);

  gemm_qkv<<<dim3(B_ * N_ / 256, 3 * D_ / 128), 256, 0, stream>>>(xb, wqb, qb, kbuf, vbuf);

  kv_kernel<<<B_ * H_, 256, 0, stream>>>(kbuf, vbuf, kvb);
  ksum_kernel<<<B_ * H_, 64, 0, stream>>>(kbuf, ksum);
  z_kernel<<<B_ * N_ * H_ / 256, 256, 0, stream>>>(qb, ksum, zbuf);
  attn_kernel<<<dim3(N_ / 128, B_ * H_), 256, 0, stream>>>(qb, kvb, zbuf, attn);

  gemm_out<<<dim3(B_ * N_ / 256, D_ / 128), 256, 0, stream>>>(attn, wob, b_out, out);
}
